// AllostericGNNLayer_25254407700535
// MI455X (gfx1250) — matrix operand order repaired and checked by reading, not compiled, not run on hardware
//
#include <hip/hip_runtime.h>
#include <hip/hip_bf16.h>

typedef __attribute__((ext_vector_type(16))) _Float16 v16h;
typedef __attribute__((ext_vector_type(8)))  _Float16 v8h;
typedef __attribute__((ext_vector_type(8)))  float    v8f;

#define HID 128
#define NH  8
#define HD  16
#define FFN 512
#define LN_EPS 1e-5f

// ---------------- WMMA fragment loaders (per ISA 16-bit layouts) ----------------
// A 16x32 f16: lanes 0-15 hold M=0..15 with K = {0..7, 16..23}; lanes 16-31 K = {8..15, 24..31}
__device__ __forceinline__ v16h load_fragA(const _Float16* __restrict__ base, int row, int stride,
                                           int k0, int khalf /*0 or 8*/) {
  const _Float16* p = base + (size_t)row * stride + k0 + khalf;
  v8h lo = *reinterpret_cast<const v8h*>(p);        // K = k0+khalf .. +7
  v8h hi = *reinterpret_cast<const v8h*>(p + 16);   // K = k0+16+khalf .. +7
  v16h r;
#pragma unroll
  for (int j = 0; j < 8; ++j) { r[j] = lo[j]; r[8 + j] = hi[j]; }
  return r;
}
// B 32x16 f16 (fed from transposed weight Wt[OUT][IN]): lane n=lane&15 holds column n,
// lanes 0-15: K = {0..7, 16..23}, lanes 16-31: K = {8..15, 24..31} (same as A)
__device__ __forceinline__ v16h load_fragB(const _Float16* __restrict__ Wt, int row, int stride,
                                           int k0, int half16 /*0 or 1*/) {
  const _Float16* p = Wt + (size_t)row * stride + k0 + half16 * 8;
  v8h lo = *reinterpret_cast<const v8h*>(p);
  v8h hi = *reinterpret_cast<const v8h*>(p + 16);
  v16h r;
#pragma unroll
  for (int j = 0; j < 8; ++j) { r[j] = lo[j]; r[8 + j] = hi[j]; }
  return r;
}

// ---------------- Tiled WMMA GEMM: out = epilogue(A[N,IN] @ Wt[OUT,IN]^T + bias) ----------------
template<int IN, int OUT, bool RES, bool GELU, bool OF32, bool OF16>
__global__ __launch_bounds__(256) void gemm_wmma_kernel(
    const _Float16* __restrict__ A, const _Float16* __restrict__ Wt,
    const float* __restrict__ bias, const float* __restrict__ res,
    float* __restrict__ out32, _Float16* __restrict__ out16, int nRows)
{
  const int colTiles = OUT / 16;
  const int rowTiles = (nRows + 15) >> 4;
  int tile = blockIdx.x * 8 + (threadIdx.x >> 5);
  if (tile >= rowTiles * colTiles) return;          // wave-uniform exit (EXEC stays all-ones)
  int rowBase = (tile / colTiles) * 16;
  int colBase = (tile % colTiles) * 16;
  int lane   = threadIdx.x & 31;
  int half16 = lane >> 4;
  int khalf  = half16 * 8;
  int idx15  = lane & 15;
  int rowA = rowBase + idx15; if (rowA >= nRows) rowA = nRows - 1;   // clamp OOB reads
  int rowB = colBase + idx15;                                        // row of Wt = out column
  v8f acc = {};
#pragma unroll
  for (int k0 = 0; k0 < IN; k0 += 32) {
    v16h a = load_fragA(A,  rowA, IN, k0, khalf);
    v16h b = load_fragB(Wt, rowB, IN, k0, half16);
    acc = __builtin_amdgcn_wmma_f32_16x16x32_f16(false, a, false, b, (short)0, acc, false, false);
  }
  int col = colBase + idx15;
  float bc = bias[col];
  float val[8];
#pragma unroll
  for (int g = 0; g < 8; ++g) {
    float t = acc[g] + bc;
    if (GELU) t = 0.5f * t * (1.0f + erff(t * 0.70710678118654752f));
    val[g] = t;
  }
  size_t base = (size_t)(rowBase + half16 * 8) * OUT + col;   // C/D: VGPR g -> M=g (+8 hi lanes)
  if (rowBase + 16 <= nRows) {                                 // wave-uniform: full tile (always
#pragma unroll                                                 // taken when nRows % 16 == 0)
    for (int g = 0; g < 8; ++g) {
      float t = val[g];
      if (RES)  t += res[base + (size_t)g * OUT];
      if (OF32) out32[base + (size_t)g * OUT] = t;
      if (OF16) out16[base + (size_t)g * OUT] = (_Float16)t;
    }
  } else {                                                     // ragged tail tile
#pragma unroll
    for (int g = 0; g < 8; ++g) {
      int row = rowBase + half16 * 8 + g;
      if (row >= nRows) continue;
      float t = val[g];
      if (RES)  t += res[base + (size_t)g * OUT];
      if (OF32) out32[base + (size_t)g * OUT] = t;
      if (OF16) out16[base + (size_t)g * OUT] = (_Float16)t;
    }
  }
}

// ---------------- LayerNorm: one wave (32 lanes) per node, 4 cols/lane ----------------
__global__ __launch_bounds__(256) void ln_kernel(const float* __restrict__ x,
                                                 const float* __restrict__ g,
                                                 const float* __restrict__ b,
                                                 _Float16* __restrict__ h16, int N)
{
  int node = blockIdx.x * 8 + (threadIdx.x >> 5);
  if (node >= N) return;
  int lane = threadIdx.x & 31;
  const float* xp = x + (size_t)node * HID;
  float v0[4]; float s = 0.f, s2 = 0.f;
#pragma unroll
  for (int j = 0; j < 4; ++j) { float t = xp[lane + 32 * j]; v0[j] = t; s += t; s2 += t * t; }
#pragma unroll
  for (int o = 16; o > 0; o >>= 1) { s += __shfl_xor(s, o, 32); s2 += __shfl_xor(s2, o, 32); }
  float mean = s * (1.0f / HID);
  float var  = s2 * (1.0f / HID) - mean * mean;
  float inv  = rsqrtf(var + LN_EPS);
#pragma unroll
  for (int j = 0; j < 4; ++j) {
    int c = lane + 32 * j;
    h16[(size_t)node * HID + c] = (_Float16)((v0[j] - mean) * inv * g[c] + b[c]);
  }
}

// ---------------- Edge pipeline ----------------
__device__ __forceinline__ unsigned f2key(float f) {
  unsigned b = __float_as_uint(f);
  return (b & 0x80000000u) ? ~b : (b | 0x80000000u);   // monotone order-preserving encode
}
__device__ __forceinline__ float key2f(unsigned k) {
  unsigned b = (k & 0x80000000u) ? (k & 0x7FFFFFFFu) : ~k;
  return __uint_as_float(b);
}

__global__ void deg_kernel(const int* __restrict__ dst, int* __restrict__ deg, int E) {
  int i = blockIdx.x * 256 + threadIdx.x;
  if (i < E) atomicAdd(&deg[dst[i]], 1);
}

__global__ void score_kernel(const int* __restrict__ src, const int* __restrict__ dst,
                             const float* __restrict__ q, const float* __restrict__ k,
                             float* __restrict__ esc, unsigned* __restrict__ mkey, int E, int N)
{
  int idx = blockIdx.x * 256 + threadIdx.x;
  if (idx >= (E + N) * NH) return;
  int e = idx >> 3, h = idx & 7;
  int s, d;
  if (e < E) { s = src[e]; d = dst[e]; } else { s = d = e - E; }
  const float* qp = q + (size_t)d * HID + h * HD;
  const float* kp = k + (size_t)s * HID + h * HD;
  float acc = 0.f;
#pragma unroll
  for (int i = 0; i < HD; ++i) acc += qp[i] * kp[i];
  acc *= 0.25f;                                   // 1/sqrt(16)
  esc[idx] = acc;
  atomicMax(&mkey[d * NH + h], f2key(acc));
}

__global__ void softmax_p_kernel(const int* __restrict__ dst, const unsigned* __restrict__ mkey,
                                 float* __restrict__ esc, float* __restrict__ z, int E, int N)
{
  int idx = blockIdx.x * 256 + threadIdx.x;
  if (idx >= (E + N) * NH) return;
  int e = idx >> 3, h = idx & 7;
  int d = (e < E) ? dst[e] : (e - E);
  float m = key2f(mkey[d * NH + h]);
  float p = expf(esc[idx] - m);
  esc[idx] = p;
  unsafeAtomicAdd(&z[d * NH + h], p);
}

__global__ void agg_kernel(const int* __restrict__ src, const int* __restrict__ dst,
                           const float* __restrict__ esc, const float* __restrict__ z,
                           const float* __restrict__ v, float* __restrict__ agg, int E, int N)
{
  int idx = blockIdx.x * 256 + threadIdx.x;
  if (idx >= (E + N) * NH) return;
  int e = idx >> 3, h = idx & 7;
  int s, d;
  if (e < E) { s = src[e]; d = dst[e]; } else { s = d = e - E; }
  float alpha = esc[idx] / z[d * NH + h];
  const float* vp = v + (size_t)s * HID + h * HD;
  float* ap = agg + (size_t)d * HID + h * HD;
#pragma unroll
  for (int i = 0; i < HD; ++i) unsafeAtomicAdd(&ap[i], alpha * vp[i]);
}

__global__ void mask_cvt_kernel(const float* __restrict__ agg, const int* __restrict__ deg,
                                _Float16* __restrict__ agg16, int N)
{
  int idx = blockIdx.x * 256 + threadIdx.x;
  if (idx >= N * HID) return;
  int n = idx / HID;
  float val = (deg[n] > 0) ? agg[idx] : 0.f;
  agg16[idx] = (_Float16)val;
}

// Weight convert + transpose: Wt[o][i] = (f16) W[i][o]
__global__ void wcvt_kernel(const float* __restrict__ W, _Float16* __restrict__ Wt, int IN_, int OUT_) {
  int idx = blockIdx.x * 256 + threadIdx.x;
  if (idx >= IN_ * OUT_) return;
  int o = idx / IN_, i = idx - o * IN_;
  Wt[idx] = (_Float16)W[(size_t)i * OUT_ + o];
}

extern "C" void kernel_launch(void* const* d_in, const int* in_sizes, int n_in,
                              void* d_out, int out_size, void* d_ws, size_t ws_size,
                              hipStream_t stream)
{
  const float* x   = (const float*)d_in[0];
  const int*   ei  = (const int*)  d_in[1];
  const float* Wq  = (const float*)d_in[2];  const float* bq  = (const float*)d_in[3];
  const float* Wk  = (const float*)d_in[4];  const float* bk  = (const float*)d_in[5];
  const float* Wv  = (const float*)d_in[6];  const float* bv  = (const float*)d_in[7];
  const float* Wo  = (const float*)d_in[8];  const float* bo  = (const float*)d_in[9];
  const float* g1  = (const float*)d_in[10]; const float* b1  = (const float*)d_in[11];
  const float* g2  = (const float*)d_in[12]; const float* b2  = (const float*)d_in[13];
  const float* W1  = (const float*)d_in[14]; const float* bf1 = (const float*)d_in[15];
  const float* W2  = (const float*)d_in[16]; const float* bf2 = (const float*)d_in[17];

  const int N = in_sizes[0] / HID;
  const int E = in_sizes[1] / 2;
  const int* src = ei;
  const int* dst = ei + E;

  // ---- workspace bump allocator (256B aligned) ----
  char* ws = (char*)d_ws; size_t off = 0;
  auto alloc = [&](size_t bytes) -> char* {
    char* p = ws + off; off = (off + bytes + 255) & ~(size_t)255; return p;
  };
  _Float16* wq_t = (_Float16*)alloc((size_t)HID * HID * 2);
  _Float16* wk_t = (_Float16*)alloc((size_t)HID * HID * 2);
  _Float16* wv_t = (_Float16*)alloc((size_t)HID * HID * 2);
  _Float16* wo_t = (_Float16*)alloc((size_t)HID * HID * 2);
  _Float16* w1_t = (_Float16*)alloc((size_t)FFN * HID * 2);   // Wt[512][128]
  _Float16* w2_t = (_Float16*)alloc((size_t)HID * FFN * 2);   // Wt[128][512]
  _Float16* h16  = (_Float16*)alloc((size_t)N * HID * 2);
  float*    q    = (float*)alloc((size_t)N * HID * 4);
  float*    k    = (float*)alloc((size_t)N * HID * 4);
  float*    v    = (float*)alloc((size_t)N * HID * 4);
  float*    esc  = (float*)alloc((size_t)(E + N) * NH * 4);
  unsigned* mkey = (unsigned*)alloc((size_t)N * NH * 4);
  float*    z    = (float*)alloc((size_t)N * NH * 4);
  int*      deg  = (int*)alloc((size_t)N * 4);
  float*    agg  = (float*)alloc((size_t)N * HID * 4);
  _Float16* agg16= (_Float16*)alloc((size_t)N * HID * 2);
  float*    x2   = (float*)alloc((size_t)N * HID * 4);
  _Float16* act16 = (_Float16*)q;   // N*FFN*2 == 51.2MB reuses q+k (dead by FFN time)

  // ---- zero-init accumulators (graph-capturable) ----
  hipMemsetAsync(mkey, 0, (size_t)N * NH * 4, stream);
  hipMemsetAsync(z,    0, (size_t)N * NH * 4, stream);
  hipMemsetAsync(deg,  0, (size_t)N * 4,      stream);
  hipMemsetAsync(agg,  0, (size_t)N * HID * 4, stream);

  // ---- weight conversion ----
  auto cdiv = [](int a, int b) { return (a + b - 1) / b; };
  wcvt_kernel<<<cdiv(HID * HID, 256), 256, 0, stream>>>(Wq, wq_t, HID, HID);
  wcvt_kernel<<<cdiv(HID * HID, 256), 256, 0, stream>>>(Wk, wk_t, HID, HID);
  wcvt_kernel<<<cdiv(HID * HID, 256), 256, 0, stream>>>(Wv, wv_t, HID, HID);
  wcvt_kernel<<<cdiv(HID * HID, 256), 256, 0, stream>>>(Wo, wo_t, HID, HID);
  wcvt_kernel<<<cdiv(HID * FFN, 256), 256, 0, stream>>>(W1, w1_t, HID, FFN); // Wt[FFN][HID]
  wcvt_kernel<<<cdiv(FFN * HID, 256), 256, 0, stream>>>(W2, w2_t, FFN, HID); // Wt[HID][FFN]

  const int rowTiles = cdiv(N, 16);
  const int lnBlocks = cdiv(N, 8);
  const int ehTotal  = (E + N) * NH;

  // ---- LN1 -> h16 ----
  ln_kernel<<<lnBlocks, 256, 0, stream>>>(x, g1, b1, h16, N);

  // ---- QKV projections (WMMA) ----
  {
    int blocks = cdiv(rowTiles * (HID / 16), 8);
    gemm_wmma_kernel<HID, HID, false, false, true, false>
        <<<blocks, 256, 0, stream>>>(h16, wq_t, bq, nullptr, q, nullptr, N);
    gemm_wmma_kernel<HID, HID, false, false, true, false>
        <<<blocks, 256, 0, stream>>>(h16, wk_t, bk, nullptr, k, nullptr, N);
    gemm_wmma_kernel<HID, HID, false, false, true, false>
        <<<blocks, 256, 0, stream>>>(h16, wv_t, bv, nullptr, v, nullptr, N);
  }

  // ---- edge attention: deg, scores+max, exp+sum, weighted aggregate ----
  deg_kernel<<<cdiv(E, 256), 256, 0, stream>>>(dst, deg, E);
  score_kernel    <<<cdiv(ehTotal, 256), 256, 0, stream>>>(src, dst, q, k, esc, mkey, E, N);
  softmax_p_kernel<<<cdiv(ehTotal, 256), 256, 0, stream>>>(dst, mkey, esc, z, E, N);
  agg_kernel      <<<cdiv(ehTotal, 256), 256, 0, stream>>>(src, dst, esc, z, v, agg, E, N);
  mask_cvt_kernel <<<cdiv(N * HID, 256), 256, 0, stream>>>(agg, deg, agg16, N);

  // ---- output projection + residual (WMMA) -> x2 ----
  gemm_wmma_kernel<HID, HID, true, false, true, false>
      <<<cdiv(rowTiles * (HID / 16), 8), 256, 0, stream>>>(agg16, wo_t, bo, x, x2, nullptr, N);

  // ---- LN2 -> h16 (reuse) ----
  ln_kernel<<<lnBlocks, 256, 0, stream>>>(x2, g2, b2, h16, N);

  // ---- FFN: gelu(h@W1+bf1) -> act16, then @W2+bf2 + x2 -> out (WMMA) ----
  gemm_wmma_kernel<HID, FFN, false, true, false, true>
      <<<cdiv(rowTiles * (FFN / 16), 8), 256, 0, stream>>>(h16, w1_t, bf1, nullptr, nullptr, act16, N);
  gemm_wmma_kernel<FFN, HID, true, false, true, false>
      <<<cdiv(rowTiles * (HID / 16), 8), 256, 0, stream>>>(act16, w2_t, bf2, x2, (float*)d_out, nullptr, N);
}
